// GraphModule_73443940762194
// MI455X (gfx1250) — compile-verified, matrix operand order checked by reading, not run on hardware
//
#include <hip/hip_runtime.h>
#include <stdint.h>

// Model dims
#define S_    1536
#define B_    2
#define D_    768
#define H_    24
#define DH_   32
#define FF_   3072
#define NTOK  (S_ * B_)          // 3072 tokens
#define SCALE_ 0.17677669529663687f   // 1/sqrt(32)

typedef __attribute__((ext_vector_type(16))) __bf16 bf16x16;
typedef __attribute__((ext_vector_type(8)))  float  f32x8;
typedef unsigned int u32x4 __attribute__((ext_vector_type(4)));
typedef int          i32x4 __attribute__((ext_vector_type(4)));
typedef int          i32x8 __attribute__((ext_vector_type(8)));

union Frag { uint32_t u[8]; bf16x16 v; };

__device__ __forceinline__ unsigned short f2bf(float f) {
  union { float f; uint32_t u; } x; x.f = f;
  uint32_t u = x.u;
  return (unsigned short)((u + 0x7FFFu + ((u >> 16) & 1u)) >> 16);
}

// K index inside a 16-bit 16x32 A-fragment (and matching 32x16 B-fragment):
// VGPR i<4 -> K = 2i (+8 for upper lane half), i>=4 -> K = 16 + 2(i-4) (+8)
__device__ __forceinline__ int frag_k(int i, int h) {
  return ((i & 3) << 1) + ((i >> 2) << 4) + (h << 3);
}

// Load a 16x32 bf16 fragment where per-lane row r = lane&15 is contiguous in K.
// K-pairs are even-aligned -> one 32-bit load per fragment register.
// Requires base and (stride*2 bytes) 4B-aligned.
__device__ __forceinline__ void load_frag_rows32(const unsigned short* base, int stride,
                                                 uint32_t out[8]) {
  int lane = threadIdx.x & 31;
  int r = lane & 15, h = lane >> 4;
  const uint32_t* row = (const uint32_t*)(base + (size_t)r * stride);
#pragma unroll
  for (int i = 0; i < 8; ++i) out[i] = row[frag_k(i, h) >> 1];
}

// --------------------------------------------------------------- TDM tile load
// Async-load a 128-row x 32-col bf16 tile (rows of 64B) from global into LDS,
// padding +16B after each row -> LDS row stride 80B (= [128][40] u16).
// Descriptor per CDNA5 ISA 8.3/8.4 (D# group0/group1), 2D tensor, groups 2/3 zero.
__device__ __forceinline__ void tdm_load_tile(uint32_t lds_off,
                                              const unsigned short* gptr,
                                              int K_elems) {
  uint64_t ga = (uint64_t)(uintptr_t)gptr;
  u32x4 g0;
  g0.x = 1u;                                                  // count=1, user desc
  g0.y = lds_off;                                             // lds_addr (bytes)
  g0.z = (uint32_t)ga;                                        // global_addr[31:0]
  g0.w = (uint32_t)((ga >> 32) & 0x01FFFFFFu) | (2u << 30);   // addr[56:32] | type=2
  i32x8 g1;
  g1[0] = (1 << 16)        // data_size = 2 bytes
        | (1 << 20)        // pad_enable
        | (3 << 22)        // pad_interval = 16 DWORDs (one 64B tile row)
        | (3 << 25);       // pad_amount  = 4 DWORDs (16B)
  g1[1] = (int)(((uint32_t)K_elems & 0xFFFFu) << 16);            // tensor_dim0[15:0]
  g1[2] = (int)(((uint32_t)K_elems >> 16) | (128u << 16));       // dim0 hi | tensor_dim1 lo
  g1[3] = (int)(32u << 16);                                      // dim1 hi | tile_dim0=32
  g1[4] = 128;                                                   // tile_dim1=128, tile_dim2=0
  g1[5] = K_elems;                                               // tensor_dim0_stride lo
  g1[6] = 0;                                                     // stride hi | dim1_stride lo
  g1[7] = 0;
  i32x4 gz = {0, 0, 0, 0};
#if defined(__clang_major__) && (__clang_major__ >= 23)
  i32x8 gz8 = {0, 0, 0, 0, 0, 0, 0, 0};
  __builtin_amdgcn_tensor_load_to_lds(g0, g1, gz, gz, gz8, 0);
#else
  __builtin_amdgcn_tensor_load_to_lds(g0, g1, gz, gz, 0);
#endif
}

// ---------------------------------------------------------------- fp32 -> bf16
__global__ __launch_bounds__(256) void cvt_kernel(const float* __restrict__ in,
                                                  unsigned short* __restrict__ out, int n) {
  int i = blockIdx.x * 256 + threadIdx.x;
  if (i < n) out[i] = f2bf(in[i]);
}

// ------------------------------------------------------------------- embedding
__global__ __launch_bounds__(256) void embed_kernel(const int* __restrict__ ids,
                                                    const float* __restrict__ wte,
                                                    const float* __restrict__ wpe,
                                                    float* __restrict__ x) {
  int t = blockIdx.x;              // token = s*B + b
  int s = t / B_, b = t % B_;
  int id = ids[b * S_ + s];
  const float* te = wte + (size_t)id * D_;
  const float* pe = wpe + (size_t)s * D_;
  float* out = x + (size_t)t * D_;
#pragma unroll
  for (int j = 0; j < 3; ++j) {
    int d = threadIdx.x + j * 256;
    out[d] = te[d] + pe[d];
  }
}

// ------------------------------------------------------------------- layernorm
template <int F32OUT>
__global__ __launch_bounds__(256) void ln_kernel(const float* __restrict__ x,
                                                 const float* __restrict__ w,
                                                 const float* __restrict__ bpar,
                                                 float* __restrict__ outF,
                                                 unsigned short* __restrict__ outB) {
  __shared__ float red[256];
  int t = blockIdx.x;
  const float* row = x + (size_t)t * D_;
  float l0 = row[threadIdx.x], l1 = row[threadIdx.x + 256], l2 = row[threadIdx.x + 512];
  red[threadIdx.x] = l0 + l1 + l2;
  __syncthreads();
  for (int off = 128; off > 0; off >>= 1) {
    if (threadIdx.x < off) red[threadIdx.x] += red[threadIdx.x + off];
    __syncthreads();
  }
  float mu = red[0] * (1.0f / 768.0f);
  __syncthreads();
  float d0 = l0 - mu, d1 = l1 - mu, d2 = l2 - mu;
  red[threadIdx.x] = d0 * d0 + d1 * d1 + d2 * d2;
  __syncthreads();
  for (int off = 128; off > 0; off >>= 1) {
    if (threadIdx.x < off) red[threadIdx.x] += red[threadIdx.x + off];
    __syncthreads();
  }
  float rstd = rsqrtf(red[0] * (1.0f / 768.0f) + 1e-5f);
  float dv[3] = {d0, d1, d2};
#pragma unroll
  for (int j = 0; j < 3; ++j) {
    int d = threadIdx.x + j * 256;
    float val = dv[j] * rstd * w[d] + bpar[d];
    if (F32OUT) outF[(size_t)t * D_ + d] = val;
    else        outB[(size_t)t * D_ + d] = f2bf(val);
  }
}

// ---------------------------------------------------------------- bf16 GEMM
// Y[n][m] = sum_k X[n][k]*W[m][k] (+bias, +resid, gelu); macro tile 128x128,
// 8 waves x (32 rows x 64 cols) = 2x4 WMMA tiles each. K staged 32-wide in LDS
// by the Tensor Data Mover (wave 0 issues, double-buffered, TENSORcnt-synced).
template <int ACT, int RES, int OUTF32>
__global__ __launch_bounds__(256) void gemm_kernel(const unsigned short* __restrict__ X,
                                                   const unsigned short* __restrict__ W,
                                                   const float* __restrict__ bias, int K,
                                                   float* __restrict__ outF,
                                                   unsigned short* __restrict__ outB,
                                                   const float* __restrict__ resid, int M) {
  __shared__ __align__(16) unsigned short At[2][128][40];
  __shared__ __align__(16) unsigned short Wt[2][128][40];
  const int n0 = blockIdx.x * 128;
  const int m0 = blockIdx.y * 128;
  const int t = threadIdx.x;
  const int wave = t >> 5;
  const int wr = wave & 3;   // row block (32 rows)
  const int wc = wave >> 2;  // col block (64 cols)
  const int nsteps = K >> 5;

  const unsigned short* Xb = X + (size_t)n0 * K;
  const unsigned short* Wb = W + (size_t)m0 * K;

  if (wave == 0) {  // prologue: stage k-chunk 0 into buffer 0
    tdm_load_tile((uint32_t)(uintptr_t)&At[0][0][0], Xb, K);
    tdm_load_tile((uint32_t)(uintptr_t)&Wt[0][0][0], Wb, K);
  }

  f32x8 acc[2][4];
  f32x8 zero = {0, 0, 0, 0, 0, 0, 0, 0};
#pragma unroll
  for (int i = 0; i < 2; ++i)
#pragma unroll
    for (int j = 0; j < 4; ++j) acc[i][j] = zero;

  for (int i = 0; i < nsteps; ++i) {
    int cur = i & 1;
    __syncthreads();  // everyone done reading the buffer TDM(i+1) will overwrite
    if (wave == 0) {
      if (i + 1 < nsteps) {
        int nxt = (i + 1) & 1;
        tdm_load_tile((uint32_t)(uintptr_t)&At[nxt][0][0], Xb + (i + 1) * 32, K);
        tdm_load_tile((uint32_t)(uintptr_t)&Wt[nxt][0][0], Wb + (i + 1) * 32, K);
        __builtin_amdgcn_s_wait_tensorcnt(2);  // stage i landed; stage i+1 in flight
      } else {
        __builtin_amdgcn_s_wait_tensorcnt(0);
      }
    }
    __syncthreads();  // LDS tile i visible to all waves

    Frag a[2], b[4];
#pragma unroll
    for (int mt = 0; mt < 2; ++mt)
      load_frag_rows32(&At[cur][wr * 32 + mt * 16][0], 40, a[mt].u);
#pragma unroll
    for (int nt = 0; nt < 4; ++nt)
      load_frag_rows32(&Wt[cur][wc * 64 + nt * 16][0], 40, b[nt].u);
#pragma unroll
    for (int mt = 0; mt < 2; ++mt)
#pragma unroll
      for (int nt = 0; nt < 4; ++nt)
        acc[mt][nt] = __builtin_amdgcn_wmma_f32_16x16x32_bf16(
            false, a[mt].v, false, b[nt].v, (short)0, acc[mt][nt], false, false);
  }

  // epilogue: C tile layout -> lane&15 = column, row = vgpr + 8*(lane>=16)
  int lane = t & 31, r = lane & 15, h = lane >> 4;
#pragma unroll
  for (int mt = 0; mt < 2; ++mt) {
#pragma unroll
    for (int nt = 0; nt < 4; ++nt) {
      int col = m0 + wc * 64 + nt * 16 + r;
      float bv = bias[col];
#pragma unroll
      for (int e = 0; e < 8; ++e) {
        int rowg = n0 + wr * 32 + mt * 16 + e + 8 * h;
        float v = acc[mt][nt][e] + bv;
        if (RES) v += resid[(size_t)rowg * M + col];
        if (ACT) v = 0.5f * v * (1.0f + erff(v * 0.70710678118654752f));
        if (OUTF32) outF[(size_t)rowg * M + col] = v;
        else        outB[(size_t)rowg * M + col] = f2bf(v);
      }
    }
  }
}

// ---------------------------------------------------------------- qkv split
// qkv[t][2304] -> q/k/v [(b*H+h)*S+s][32] bf16
__global__ __launch_bounds__(256) void split_qkv_kernel(const unsigned short* __restrict__ qkv,
                                                        unsigned short* __restrict__ q,
                                                        unsigned short* __restrict__ k,
                                                        unsigned short* __restrict__ v) {
  int i = blockIdx.x * 256 + threadIdx.x;
  if (i >= NTOK * 3 * D_) return;
  int tok = i / (3 * D_), c = i % (3 * D_);
  int h = c / 96, rem = c % 96, which = rem / 32, dh = rem % 32;
  int s = tok / B_, b = tok % B_;
  size_t di = ((size_t)(b * H_ + h) * S_ + s) * DH_ + dh;
  unsigned short val = qkv[i];
  if (which == 0) q[di] = val;
  else if (which == 1) k[di] = val;
  else v[di] = val;
}

// ---------------------------------------------------------------- attention
// One wave per (b, h, 16-query block). Flash-style online softmax.
__global__ __launch_bounds__(256) void attn_kernel(const unsigned short* __restrict__ Q,
                                                   const unsigned short* __restrict__ Kb,
                                                   const unsigned short* __restrict__ Vb,
                                                   unsigned short* __restrict__ ctx) {
  __shared__ __align__(16) unsigned short pS[8][16][36];  // per-wave 16x32 bf16 P scratch
  int lane = threadIdx.x & 31, w = threadIdx.x >> 5;
  int gw = blockIdx.x * 8 + w;
  int qblk = gw % (S_ / 16);
  int bh = gw / (S_ / 16);
  int b = bh / H_, h = bh % H_;
  int s0 = qblk * 16;
  const unsigned short* Qh = Q + (size_t)bh * S_ * DH_;
  const unsigned short* Kh = Kb + (size_t)bh * S_ * DH_;
  const unsigned short* Vh = Vb + (size_t)bh * S_ * DH_;
  int r = lane & 15, hh = lane >> 4;

  Frag qf;
  load_frag_rows32(Qh + (size_t)s0 * DH_, DH_, qf.u);  // 16 queries x DH=32 = one A frag

  f32x8 zero = {0, 0, 0, 0, 0, 0, 0, 0};
  f32x8 cacc0 = zero, cacc1 = zero;  // ctx: 16 x (dh 0..15 / 16..31)
  float rmax[8], rsum[8];
#pragma unroll
  for (int e = 0; e < 8; ++e) { rmax[e] = -3.0e38f; rsum[e] = 0.0f; }

  int nkb = (s0 + 16 + 31) / 32;  // causal: keys only up to s0+15
  for (int kblk = 0; kblk < nkb; ++kblk) {
    int key0 = kblk * 32;
    Frag kf0, kf1;
    load_frag_rows32(Kh + (size_t)key0 * DH_, DH_, kf0.u);         // keys key0..+15 as B cols
    load_frag_rows32(Kh + (size_t)(key0 + 16) * DH_, DH_, kf1.u);  // keys +16..+31
    f32x8 sc0 = __builtin_amdgcn_wmma_f32_16x16x32_bf16(false, qf.v, false, kf0.v,
                                                        (short)0, zero, false, false);
    f32x8 sc1 = __builtin_amdgcn_wmma_f32_16x16x32_bf16(false, qf.v, false, kf1.v,
                                                        (short)0, zero, false, false);
#pragma unroll
    for (int e = 0; e < 8; ++e) {
      int qs = s0 + e + 8 * hh;
      float v0 = sc0[e] * SCALE_; if (key0 + r > qs)      v0 = -10000.0f;
      float v1 = sc1[e] * SCALE_; if (key0 + 16 + r > qs) v1 = -10000.0f;
      float rm = fmaxf(v0, v1);
#pragma unroll
      for (int off = 1; off < 16; off <<= 1) rm = fmaxf(rm, __shfl_xor(rm, off, 32));
      float mn = fmaxf(rmax[e], rm);
      float alpha = __expf(rmax[e] - mn);
      float p0 = __expf(v0 - mn), p1 = __expf(v1 - mn);
      float ps = p0 + p1;
#pragma unroll
      for (int off = 1; off < 16; off <<= 1) ps += __shfl_xor(ps, off, 32);
      rsum[e] = rsum[e] * alpha + ps;
      rmax[e] = mn;
      cacc0[e] *= alpha;
      cacc1[e] *= alpha;
      pS[w][e + 8 * hh][r]      = f2bf(p0);  // row = query, col = key-in-block
      pS[w][e + 8 * hh][16 + r] = f2bf(p1);
    }
    // C-layout P -> A-fragment (wave-private LDS; per-wave LDS ops are in-order)
    Frag pf;
    {
      const uint32_t* prow = (const uint32_t*)&pS[w][r][0];
#pragma unroll
      for (int i = 0; i < 8; ++i) pf.u[i] = prow[frag_k(i, hh) >> 1];
    }
    // V as B: B[k=key][n=dh]; pairs stride one key row (DH_ shorts)
    Frag vf0, vf1;
#pragma unroll
    for (int i = 0; i < 8; ++i) {
      int k = frag_k(i, hh);
      const unsigned short* vr0 = Vh + (size_t)(key0 + k) * DH_;
      const unsigned short* vr1 = Vh + (size_t)(key0 + k + 1) * DH_;
      vf0.u[i] = (uint32_t)vr0[r]      | ((uint32_t)vr1[r] << 16);
      vf1.u[i] = (uint32_t)vr0[16 + r] | ((uint32_t)vr1[16 + r] << 16);
    }
    cacc0 = __builtin_amdgcn_wmma_f32_16x16x32_bf16(false, pf.v, false, vf0.v,
                                                    (short)0, cacc0, false, false);
    cacc1 = __builtin_amdgcn_wmma_f32_16x16x32_bf16(false, pf.v, false, vf1.v,
                                                    (short)0, cacc1, false, false);
  }

  // epilogue: ctx[t][h*32+dh] bf16, token-major for the WO GEMM
#pragma unroll
  for (int e = 0; e < 8; ++e) {
    int s = s0 + e + 8 * hh;
    float inv = 1.0f / rsum[e];
    size_t base = (size_t)(s * B_ + b) * D_ + h * DH_;
    ctx[base + r]      = f2bf(cacc0[e] * inv);
    ctx[base + 16 + r] = f2bf(cacc1[e] * inv);
  }
}

// =============================================================== host driver
extern "C" void kernel_launch(void* const* d_in, const int* in_sizes, int n_in,
                              void* d_out, int out_size, void* d_ws, size_t ws_size,
                              hipStream_t stream) {
  (void)in_sizes; (void)n_in; (void)out_size; (void)ws_size;
  const int*   ids   = (const int*)d_in[0];
  const float* wte   = (const float*)d_in[3];
  const float* wpe   = (const float*)d_in[4];
  const float* lnf_w = (const float*)d_in[29];
  const float* lnf_b = (const float*)d_in[30];

  char* wsp = (char*)d_ws;
  auto alloc = [&](size_t bytes) -> void* {
    void* p = (void*)wsp;
    wsp += (bytes + 255) & ~((size_t)255);
    return p;
  };

  float*          x    = (float*)alloc((size_t)NTOK * D_ * 4);
  unsigned short* hb   = (unsigned short*)alloc((size_t)NTOK * D_ * 2);
  unsigned short* qkvb = (unsigned short*)alloc((size_t)NTOK * 3 * D_ * 2);
  unsigned short* qb   = (unsigned short*)alloc((size_t)B_ * H_ * S_ * DH_ * 2);
  unsigned short* kb   = (unsigned short*)alloc((size_t)B_ * H_ * S_ * DH_ * 2);
  unsigned short* vb   = (unsigned short*)alloc((size_t)B_ * H_ * S_ * DH_ * 2);
  unsigned short* ctxb = (unsigned short*)alloc((size_t)NTOK * D_ * 2);
  unsigned short* actb = (unsigned short*)alloc((size_t)NTOK * FF_ * 2);
  unsigned short* wbf[2][4];
  const int wsz[4] = {3 * D_ * D_, D_ * D_, FF_ * D_, D_ * FF_};
  for (int l = 0; l < 2; ++l)
    for (int j = 0; j < 4; ++j)
      wbf[l][j] = (unsigned short*)alloc((size_t)wsz[j] * 2);

  // weight conversion fp32 -> bf16 (wqkv, wo, wfc, wproj per layer)
  for (int l = 0; l < 2; ++l) {
    int base = 5 + l * 12;
    const int widx[4] = {base + 2, base + 4, base + 8, base + 10};
    for (int j = 0; j < 4; ++j)
      cvt_kernel<<<(wsz[j] + 255) / 256, 256, 0, stream>>>((const float*)d_in[widx[j]],
                                                           wbf[l][j], wsz[j]);
  }

  // embedding -> residual stream x[(s*B+b)][D]
  embed_kernel<<<NTOK, 256, 0, stream>>>(ids, wte, wpe, x);

  for (int l = 0; l < 2; ++l) {
    int base = 5 + l * 12;
    const float* ln1w  = (const float*)d_in[base + 0];
    const float* ln1b  = (const float*)d_in[base + 1];
    const float* bqkv  = (const float*)d_in[base + 3];
    const float* bo    = (const float*)d_in[base + 5];
    const float* ln2w  = (const float*)d_in[base + 6];
    const float* ln2b  = (const float*)d_in[base + 7];
    const float* bfc   = (const float*)d_in[base + 9];
    const float* bproj = (const float*)d_in[base + 11];

    // h = LN1(x) (bf16)
    ln_kernel<0><<<NTOK, 256, 0, stream>>>(x, ln1w, ln1b, nullptr, hb);
    // qkv = h @ wqkv^T + bqkv  (3072x768 @ 768x2304)
    gemm_kernel<0, 0, 0><<<dim3(NTOK / 128, (3 * D_) / 128), 256, 0, stream>>>(
        hb, wbf[l][0], bqkv, D_, nullptr, qkvb, nullptr, 3 * D_);
    split_qkv_kernel<<<(NTOK * 3 * D_) / 256, 256, 0, stream>>>(qkvb, qb, kb, vb);
    // flash attention: B*H*(S/16) = 4608 waves / 8 per block
    attn_kernel<<<(B_ * H_ * (S_ / 16)) / 8, 256, 0, stream>>>(qb, kb, vb, ctxb);
    // x = x + ctx @ wo^T + bo  (in-place residual)
    gemm_kernel<0, 1, 1><<<dim3(NTOK / 128, D_ / 128), 256, 0, stream>>>(
        ctxb, wbf[l][1], bo, D_, x, nullptr, x, D_);
    // h = LN2(x)
    ln_kernel<0><<<NTOK, 256, 0, stream>>>(x, ln2w, ln2b, nullptr, hb);
    // act = gelu(h @ wfc^T + bfc)  (3072x768 @ 768x3072)
    gemm_kernel<1, 0, 0><<<dim3(NTOK / 128, FF_ / 128), 256, 0, stream>>>(
        hb, wbf[l][2], bfc, D_, nullptr, actb, nullptr, FF_);
    // x = x + act @ wproj^T + bproj  (K = 3072)
    gemm_kernel<0, 1, 1><<<dim3(NTOK / 128, D_ / 128), 256, 0, stream>>>(
        actb, wbf[l][3], bproj, FF_, x, nullptr, x, D_);
  }

  // final LN -> fp32 output
  ln_kernel<1><<<NTOK, 256, 0, stream>>>(x, lnf_w, lnf_b, (float*)d_out, nullptr);
}
